// CapsuleLayer_23716809408731
// MI455X (gfx1250) — matrix operand order repaired and checked by reading, not compiled, not run on hardware
//
#include <hip/hip_runtime.h>

// CapsuleLayer dynamic routing for MI455X (gfx1250, wave32, WMMA).
// W -> bf16 once in TWO layouts (row [i,j,p,d] for R1-A, transposed [i,j,d,p]
// for R2-B), both L2-resident (128MB < 192MB L2). u_hat is never materialized:
// each routing pass recomputes it with v_wmma_f32_16x16x32_bf16.
// Needs ~142.2 MB of d_ws scratch.

#define B_ 32
#define I_ 2048
#define J_ 32
#define P_ 16
#define D_ 32

typedef __attribute__((ext_vector_type(8)))  float          v8f;
typedef __attribute__((ext_vector_type(16))) __bf16         v16bf;
typedef __attribute__((ext_vector_type(4)))  float          f32x4;
typedef __attribute__((ext_vector_type(4)))  unsigned int   u32x4;
typedef __attribute__((ext_vector_type(4)))  unsigned short u16x4;

union FragBF { v16bf v; unsigned int w[8]; unsigned short u[16]; };
union XQ     { u32x4 q; unsigned int w[4]; unsigned short u[8];  };

__device__ __forceinline__ unsigned short f2bf(float f) {
  unsigned int x = __float_as_uint(f);
  x += 0x7FFFu + ((x >> 16) & 1u);          // round-to-nearest-even
  return (unsigned short)(x >> 16);
}
__device__ __forceinline__ float bf2f(unsigned short h) {
  return __uint_as_float(((unsigned int)h) << 16);
}
// packed bf16 multiply: one VOP3P op handles two A-fragment elements
__device__ __forceinline__ unsigned int pk_mul_bf16(unsigned int a, unsigned int b) {
  unsigned int d;
  asm("v_pk_mul_bf16 %0, %1, %2" : "=v"(d) : "v"(a), "v"(b));
  return d;
}

// ---------------------------------------------------------------- converts
__global__ void caps_cvt_kernel(const float* __restrict__ src,
                                unsigned short* __restrict__ dst, int n4) {
  int i = blockIdx.x * blockDim.x + threadIdx.x;
  const int stride = gridDim.x * blockDim.x;
  for (; i < n4; i += stride) {
    f32x4 f = *(const f32x4*)(const void*)(src + 4 * (size_t)i);
    u16x4 o;
    o[0] = f2bf(f[0]); o[1] = f2bf(f[1]); o[2] = f2bf(f[2]); o[3] = f2bf(f[3]);
    *(u16x4*)(void*)(dst + 4 * (size_t)i) = o;
  }
}

// Wt[i,j,d,p] = bf16(W[i,j,p,d]).  thread -> (ij, d): reads coalesced across
// lanes (consecutive d), writes one contiguous 32B run of 16 p's.
__global__ __launch_bounds__(256)
void caps_wt_kernel(const float* __restrict__ W, unsigned short* __restrict__ Wt) {
  const int t  = blockIdx.x * 256 + threadIdx.x;   // exactly I_*J_*D_ threads
  const int d  = t & 31;
  const int ij = t >> 5;
  const float* src = W + (size_t)ij * P_ * D_ + d;
  unsigned int w[8];
#pragma unroll
  for (int k = 0; k < 8; ++k) {
    const unsigned short lo = f2bf(src[(size_t)(2 * k) * D_]);
    const unsigned short hs = f2bf(src[(size_t)(2 * k + 1) * D_]);
    w[k] = (unsigned int)lo | ((unsigned int)hs << 16);
  }
  u32x4 o0 = {w[0], w[1], w[2], w[3]};
  u32x4 o1 = {w[4], w[5], w[6], w[7]};
  unsigned short* dst = Wt + ((size_t)ij * D_ + d) * P_;
  *(u32x4*)(void*)dst = o0;
  *(u32x4*)(void*)(dst + 8) = o1;
}

__global__ void caps_zero_kernel(float* __restrict__ p, int n) {
  int i = blockIdx.x * blockDim.x + threadIdx.x;
  const int stride = gridDim.x * blockDim.x;
  for (; i < n; i += stride) p[i] = 0.f;
}

// ---------------------------------------------------------------- A-fragment (R2)
// 16-bit A (16x32): lane<16 (M=lane): e<8 -> K=e ; e>=8 -> K=e+8
//                   lane>=16 (M=lane-16): e<8 -> K=e+8 ; e>=8 -> K=e+16
// K = pair*16+p: K<16 -> capsule i0, K>=16 -> capsule i1.  c folded in via
// packed bf16 multiplies.
__device__ __forceinline__ FragBF r2_build_a(const unsigned short* __restrict__ Xb,
                                             const unsigned short* __restrict__ cb,
                                             unsigned int uc_pk, int b,
                                             int i0, int i1, int j, int pbase) {
  XQ x0, x1;
  x0.q = *(const u32x4*)(const void*)(Xb + ((size_t)b * I_ + i0) * P_ + pbase);
  x1.q = *(const u32x4*)(const void*)(Xb + ((size_t)b * I_ + i1) * P_ + pbase);
  unsigned int cp0, cp1;
  if (uc_pk != 0u) {
    cp0 = uc_pk; cp1 = uc_pk;
  } else {
    cp0 = 0x10001u * (unsigned int)cb[((size_t)b * I_ + i0) * J_ + j];
    cp1 = 0x10001u * (unsigned int)cb[((size_t)b * I_ + i1) * J_ + j];
  }
  FragBF a;
#pragma unroll
  for (int k = 0; k < 4; ++k) {
    a.w[k]     = pk_mul_bf16(x0.w[k], cp0);   // K in [0,16): capsule i0
    a.w[k + 4] = pk_mul_bf16(x1.w[k], cp1);   // K in [16,32): capsule i1
  }
  return a;
}

// ---------------------------------------------------------------- R2: s GEMM
// s[b,j,d] = sum_{i,p} (c[b,i,j]*x[b,i,p]) * W[i,j,p,d]
// Per j: GEMM M=B=32, N=D=32, K=I*P=32768; each K=32 step covers 2 capsules.
// B-fragments come from transposed Wt in 2x b128 loads per fragment.
__global__ __launch_bounds__(128)
void caps_r2_kernel(const unsigned short* __restrict__ Xb,
                    const unsigned short* __restrict__ Wt,
                    const unsigned short* __restrict__ cb,
                    float* __restrict__ s, unsigned int uc_pk) {
  const int j     = blockIdx.x;
  const int wave  = threadIdx.x >> 5;
  const int lane  = threadIdx.x & 31;
  const int l     = lane & 15;
  const bool hi   = lane >= 16;
  const int pbase = hi ? 8 : 0;

  v8f acc00 = {0,0,0,0,0,0,0,0}, acc01 = {0,0,0,0,0,0,0,0};
  v8f acc10 = {0,0,0,0,0,0,0,0}, acc11 = {0,0,0,0,0,0,0,0};

  const int ibase = blockIdx.y * 128 + wave * 32;
#pragma unroll 2
  for (int kk = 0; kk < 16; ++kk) {
    const int i0 = ibase + kk * 2;
    const int i1 = i0 + 1;

    // B (32x16): e<8 -> i0, p=pbase+e ; e>=8 -> i1, p=pbase+e-8 (matches A).
    // Wt layout [i,j,d,p]: 16 p's contiguous -> two b128 loads per fragment.
    const unsigned short* w0 = Wt + (((size_t)i0 * J_ + j) * D_ + l) * P_ + pbase;
    const unsigned short* w1 = Wt + (((size_t)i1 * J_ + j) * D_ + l) * P_ + pbase;
    FragBF b0, b1;
    *(u32x4*)&b0.w[0] = *(const u32x4*)(const void*)(w0);
    *(u32x4*)&b0.w[4] = *(const u32x4*)(const void*)(w1);
    *(u32x4*)&b1.w[0] = *(const u32x4*)(const void*)(w0 + 16 * P_);       // d = l+16
    *(u32x4*)&b1.w[4] = *(const u32x4*)(const void*)(w1 + 16 * P_);

    FragBF a0 = r2_build_a(Xb, cb, uc_pk, l,      i0, i1, j, pbase);
    FragBF a1 = r2_build_a(Xb, cb, uc_pk, l + 16, i0, i1, j, pbase);

    acc00 = __builtin_amdgcn_wmma_f32_16x16x32_bf16(false, a0.v, false, b0.v, (short)0, acc00, false, false);
    acc01 = __builtin_amdgcn_wmma_f32_16x16x32_bf16(false, a0.v, false, b1.v, (short)0, acc01, false, false);
    acc10 = __builtin_amdgcn_wmma_f32_16x16x32_bf16(false, a1.v, false, b0.v, (short)0, acc10, false, false);
    acc11 = __builtin_amdgcn_wmma_f32_16x16x32_bf16(false, a1.v, false, b1.v, (short)0, acc11, false, false);
  }

  // cross-wave reduce in LDS, then f32 atomics into global s
  __shared__ float sacc[1024];                // [mh][dh][vgpr r][lane]
  for (int t = threadIdx.x; t < 1024; t += 128) sacc[t] = 0.f;
  __syncthreads();
#pragma unroll
  for (int r = 0; r < 8; ++r) {
    atomicAdd(&sacc[((0 * 2 + 0) * 8 + r) * 32 + lane], acc00[r]);
    atomicAdd(&sacc[((0 * 2 + 1) * 8 + r) * 32 + lane], acc01[r]);
    atomicAdd(&sacc[((1 * 2 + 0) * 8 + r) * 32 + lane], acc10[r]);
    atomicAdd(&sacc[((1 * 2 + 1) * 8 + r) * 32 + lane], acc11[r]);
  }
  __syncthreads();
  for (int t = threadIdx.x; t < 1024; t += 128) {
    const int lidx = t & 31;
    const int r    = (t >> 5) & 7;
    const int dh   = (t >> 8) & 1;
    const int mh   = (t >> 9) & 1;
    const int b = mh * 16 + r + ((lidx >= 16) ? 8 : 0);   // C layout: hi lanes = M+8
    const int d = dh * 16 + (lidx & 15);
    atomicAdd(&s[((size_t)b * J_ + j) * D_ + d], sacc[t]);
  }
}

// ------------------------------------------------- R1: agreement + softmax
// Per (i,j): z[p,b] = W_ij(PxD) x v^T(DxB)  -- M=16, N=16x2, K=D=32 (full K).
// a[b,i,j] = sum_p x[b,i,p]*z[p,b]: C layout puts p-halves of the same b on
// lanes l and l+16 -> 8 FMAs + one shfl_xor(16). No LDS atomics needed.
__global__ __launch_bounds__(256)
void caps_r1_kernel(const unsigned short* __restrict__ Xb,
                    const unsigned short* __restrict__ Wb,
                    const unsigned short* __restrict__ vb,
                    float* __restrict__ blog, unsigned short* __restrict__ cb,
                    int first) {
  const int i    = blockIdx.x;
  const int wave = threadIdx.x >> 5;
  const int lane = threadIdx.x & 31;
  const int l    = lane & 15;
  const bool hi  = lane >= 16;
  const int pbase = hi ? 8 : 0;

  __shared__ float bu[J_ * B_];               // [j][b] (transposed: conflict-free)

  // epilogue x: lane supplies x[b'=nh*16+l, i, pbase..pbase+7]
  XQ xe0, xe1;
  xe0.q = *(const u32x4*)(const void*)(Xb + ((size_t)(l)      * I_ + i) * P_ + pbase);
  xe1.q = *(const u32x4*)(const void*)(Xb + ((size_t)(l + 16) * I_ + i) * P_ + pbase);

  for (int j = wave; j < J_; j += 8) {
    // A = W_ij (M=p, K=d): lane row p=l; two 8-wide d-runs, 2x b128 loads.
    const unsigned short* wrow = Wb + ((size_t)(i * J_ + j) * P_ + l) * D_;
    FragBF a;
    *(u32x4*)&a.w[0] = *(const u32x4*)(const void*)(wrow + pbase);
    *(u32x4*)&a.w[4] = *(const u32x4*)(const void*)(wrow + 16 + pbase);

#pragma unroll
    for (int nh = 0; nh < 2; ++nh) {
      // B = v^T (K=d 32, N=b' 16): e<8 d=pbase+e, e>=8 d=16+pbase+e-8; n=l.
      const int bp = nh * 16 + l;
      const unsigned short* vrow = vb + ((size_t)bp * J_ + j) * D_ + pbase;
      FragBF bf;
      *(u32x4*)&bf.w[0] = *(const u32x4*)(const void*)(vrow);
      *(u32x4*)&bf.w[4] = *(const u32x4*)(const void*)(vrow + 16);

      v8f acc = {0,0,0,0,0,0,0,0};
      acc = __builtin_amdgcn_wmma_f32_16x16x32_bf16(false, a.v, false, bf.v,
                                                    (short)0, acc, false, false);
      const XQ& xe = nh ? xe1 : xe0;
      float part = 0.f;
#pragma unroll
      for (int r = 0; r < 8; ++r) part += bf2f(xe.u[r]) * acc[r];   // p = r+pbase
      part += __shfl_xor(part, 16);                                 // join p-halves
      if (!hi) bu[j * B_ + bp] = part;
    }
  }
  __syncthreads();

  // logits update + softmax over j for this capsule (32 rows, one thread each)
  if (threadIdx.x < B_) {
    const int b = threadIdx.x;
    float bm = -3.402823466e38f;
    for (int j = 0; j < J_; ++j) {
      float t = bu[j * B_ + b];
      if (!first) t += blog[((size_t)b * I_ + i) * J_ + j];
      blog[((size_t)b * I_ + i) * J_ + j] = t;
      bu[j * B_ + b] = t;
      bm = fmaxf(bm, t);
    }
    float sum = 0.f;
    for (int j = 0; j < J_; ++j) {
      const float e = __expf(bu[j * B_ + b] - bm);
      bu[j * B_ + b] = e;
      sum += e;
    }
    const float inv = 1.f / sum;
    for (int j = 0; j < J_; ++j)
      cb[((size_t)b * I_ + i) * J_ + j] = f2bf(bu[j * B_ + b] * inv);
  }
}

// ---------------------------------------------------------------- squash
__global__ __launch_bounds__(32)
void caps_squash_kernel(const float* __restrict__ s,
                        unsigned short* __restrict__ vb,
                        float* __restrict__ fout) {
  const int row = blockIdx.x;                 // (b*J + j)
  const int d   = threadIdx.x;
  const float x = s[(size_t)row * D_ + d];
  float n2 = x * x;
  n2 += __shfl_xor(n2, 1);
  n2 += __shfl_xor(n2, 2);
  n2 += __shfl_xor(n2, 4);
  n2 += __shfl_xor(n2, 8);
  n2 += __shfl_xor(n2, 16);
  const float sc  = n2 / ((1.f + n2) * sqrtf(n2 + 1e-7f));
  const float val = x * sc;
  vb[(size_t)row * D_ + d] = f2bf(val);
  if (fout) fout[(size_t)row * D_ + d] = val;
}

// ---------------------------------------------------------------- launch
extern "C" void kernel_launch(void* const* d_in, const int* in_sizes, int n_in,
                              void* d_out, int out_size, void* d_ws, size_t ws_size,
                              hipStream_t stream) {
  const float* X = (const float*)d_in[0];   // [B, I, P]
  const float* W = (const float*)d_in[1];   // [1, I, J, P, D]

  // workspace layout (~142.2 MB)
  char* ws = (char*)d_ws;
  unsigned short* Wb = (unsigned short*)ws;                  // 67,108,864 B  [i,j,p,d]
  unsigned short* Wt = (unsigned short*)(ws + 67108864);     // 67,108,864 B  [i,j,d,p]
  unsigned short* Xb = (unsigned short*)(ws + 134217728);    //  2,097,152 B
  float*          blog = (float*)(ws + 136314880);           //  8,388,608 B
  unsigned short* cb = (unsigned short*)(ws + 144703488);    //  4,194,304 B (bf16 c)
  float*          s  = (float*)(ws + 148897792);             //    131,072 B
  unsigned short* vb = (unsigned short*)(ws + 149028864);    //     65,536 B (bf16 v)
  float* out = (float*)d_out;

  const unsigned int UC = 0x3D003D00u;  // packed bf16(1/32, 1/32)

  caps_cvt_kernel<<<4096, 256, 0, stream>>>(W, Wb, (I_ * J_ * P_ * D_) / 4);
  caps_wt_kernel<<<(I_ * J_ * D_) / 256, 256, 0, stream>>>(W, Wt);
  caps_cvt_kernel<<<1024, 256, 0, stream>>>(X, Xb, (B_ * I_ * P_) / 4);

  // routing iteration 0: c uniform (b logits = 0)
  caps_zero_kernel<<<128, 256, 0, stream>>>(s, B_ * J_ * D_);
  caps_r2_kernel<<<dim3(J_, 16), 128, 0, stream>>>(Xb, Wt, cb, s, UC);
  caps_squash_kernel<<<B_ * J_, 32, 0, stream>>>(s, vb, nullptr);

  // routing iteration 1
  caps_r1_kernel<<<I_, 256, 0, stream>>>(Xb, Wb, vb, blog, cb, 1);
  caps_zero_kernel<<<128, 256, 0, stream>>>(s, B_ * J_ * D_);
  caps_r2_kernel<<<dim3(J_, 16), 128, 0, stream>>>(Xb, Wt, cb, s, 0u);
  caps_squash_kernel<<<B_ * J_, 32, 0, stream>>>(s, vb, nullptr);

  // routing iteration 2 (final -> d_out)
  caps_r1_kernel<<<I_, 256, 0, stream>>>(Xb, Wb, vb, blog, cb, 0);
  caps_zero_kernel<<<128, 256, 0, stream>>>(s, B_ * J_ * D_);
  caps_r2_kernel<<<dim3(J_, 16), 128, 0, stream>>>(Xb, Wt, cb, s, 0u);
  caps_squash_kernel<<<B_ * J_, 32, 0, stream>>>(s, vb, out);
}
